// VanillaRNN_86071144612210
// MI455X (gfx1250) — matrix operand order repaired and checked by reading, not compiled, not run on hardware
//
#include <hip/hip_runtime.h>
#include <hip/hip_bf16.h>

#define B_DIM 256
#define T_DIM 512
#define H_DIM 1024
#define O_DIM 10

typedef __attribute__((ext_vector_type(16))) __bf16       v16bf;
typedef __attribute__((ext_vector_type(8)))  float        v8f;
typedef __attribute__((ext_vector_type(8)))  unsigned int v8u;

// scheduling fence: nothing may be reordered across (enforces SW pipeline)
#define SCHED_FENCE() __builtin_amdgcn_sched_barrier(0)

static __device__ __forceinline__ unsigned short f32_to_bf16_rne(float f) {
    unsigned int u = __float_as_uint(f);
    u += 0x7fffu + ((u >> 16) & 1u);           // round-to-nearest-even
    return (unsigned short)(u >> 16);
}
static __device__ __forceinline__ float bf16_to_f32(unsigned short h) {
    return __uint_as_float(((unsigned int)h) << 16);
}

// ---------------------------------------------------------------------------
// Pack W_hh (f32 row-major [H,H]) into bf16 B-fragment layout:
// tile (kt, nt) covers K=kt*32..+31, N=nt*16..+15.
// Lane L holds column N = nt*16 + (L&15), h = (L>=16),
// dword v packs W[k][n] (lo) and W[k+1][n] (hi) with k = base + 8h + 2v + (v>=4 ? 8 : 0).
// Storage: Bp[((kt*64 + nt)*32 + lane)*8 + v]  -> 32 contiguous bytes/lane.
// ---------------------------------------------------------------------------
__global__ void pack_whh_kernel(const float* __restrict__ Whh,
                                unsigned int* __restrict__ Bp) {
    unsigned int id = blockIdx.x * blockDim.x + threadIdx.x;   // 524288 total
    unsigned int v    = id & 7u;
    unsigned int lane = (id >> 3) & 31u;
    unsigned int nt   = (id >> 8) & 63u;
    unsigned int kt   = id >> 14;
    unsigned int n = nt * 16u + (lane & 15u);
    unsigned int k = kt * 32u + ((lane & 16u) ? 8u : 0u) + ((v & 4u) ? 8u : 0u) + 2u * v;
    unsigned short lo = f32_to_bf16_rne(Whh[k * H_DIM + n]);
    unsigned short hi = f32_to_bf16_rne(Whh[(k + 1u) * H_DIM + n]);
    Bp[id] = ((unsigned int)hi << 16) | (unsigned int)lo;
}

__global__ void zero_u32_kernel(unsigned int* __restrict__ p, int n) {
    int i = blockIdx.x * blockDim.x + threadIdx.x;
    if (i < n) p[i] = 0u;
}

// ---- fragment load helpers (base + compile-time immediate offsets) --------
static __device__ __forceinline__ v8u load_a_frag(const unsigned short* aptr, int kt) {
    const uint4 alo = *(const uint4*)(aptr + kt * 32);        // +64 B per kt
    const uint4 ahi = *(const uint4*)(aptr + kt * 32 + 16);
    return v8u{alo.x, alo.y, alo.z, alo.w, ahi.x, ahi.y, ahi.z, ahi.w};
}
static __device__ __forceinline__ v8u load_b_frag(const unsigned int* bptr, int kt, int j) {
    return *(const v8u*)(bptr + kt * 16384 + j * 256);        // +64 KB per kt
}

// ---------------------------------------------------------------------------
// One recurrence step:  h_next = tanh( h_prev @ W_hh + x[:,t]*W_hx + b_h )
// Each wave computes a 16(M) x 64(N) strip of the [256,1024] result.
// 256 waves total = 32 blocks x 8 waves (wave32).
// K-loop fully unrolled, software-pipelined at distance 2, with
// sched_barrier(0) fences so the backend cannot collapse the pipeline:
// WMMAs for kt consume loads issued two groups earlier -> partial loadcnt
// waits (<= 20 outstanding) instead of draining to zero.
// ---------------------------------------------------------------------------
__global__ void __launch_bounds__(256)
rnn_step_kernel(const unsigned short* __restrict__ hprev,   // bf16 [256,1024]
                const unsigned int*  __restrict__ Bp,       // packed W_hh
                const float* __restrict__ x,                // [256,512]
                const float* __restrict__ Whx,              // [1024]
                const float* __restrict__ bh,               // [1024]
                int t,
                unsigned short* __restrict__ hnext) {       // bf16 [256,1024]
    const int lane    = threadIdx.x & 31;
    const int wave    = threadIdx.x >> 5;
    const int gw      = blockIdx.x * 8 + wave;   // 0..255
    const int m_strip = gw >> 4;                 // 0..15 -> M block of 16
    const int n_strip = gw & 15;                 // 0..15 -> N block of 64
    const int m0   = m_strip * 16;
    const int lrow = lane & 15;
    const int lhi  = lane >> 4;                  // 0 or 1
    const int row  = m0 + lrow;                  // A-fragment row (M)
    const int klo  = lhi * 8;                    // A K split per ISA layout

    const unsigned short* aptr = hprev + row * H_DIM + klo;
    const unsigned int*   bptr =
        Bp + ((unsigned)(n_strip * 4) * 32u + (unsigned)lane) * 8u;

    v8f acc[4] = {v8f{}, v8f{}, v8f{}, v8f{}};

    v8u a_buf[3];
    v8u b_buf[3][4];

    // prologue: fragments for kt = 0 and kt = 1
    a_buf[0] = load_a_frag(aptr, 0);
#pragma unroll
    for (int j = 0; j < 4; ++j) b_buf[0][j] = load_b_frag(bptr, 0, j);
    SCHED_FENCE();
    a_buf[1] = load_a_frag(aptr, 1);
#pragma unroll
    for (int j = 0; j < 4; ++j) b_buf[1][j] = load_b_frag(bptr, 1, j);
    SCHED_FENCE();

#pragma unroll
    for (int kt = 0; kt < 32; ++kt) {
        const int cur = kt % 3;
        if (kt + 2 < 32) {
            const int pf = (kt + 2) % 3;
            a_buf[pf] = load_a_frag(aptr, kt + 2);
#pragma unroll
            for (int j = 0; j < 4; ++j) b_buf[pf][j] = load_b_frag(bptr, kt + 2, j);
        }
        SCHED_FENCE();
        v16bf a = __builtin_bit_cast(v16bf, a_buf[cur]);
#pragma unroll
        for (int j = 0; j < 4; ++j) {
            v16bf b = __builtin_bit_cast(v16bf, b_buf[cur][j]);
            acc[j] = __builtin_amdgcn_wmma_f32_16x16x32_bf16(
                false, a, false, b, (short)0, acc[j], false, false);
        }
        SCHED_FENCE();
    }

    // Epilogue: fuse rank-1 input projection + bias + tanh, store bf16.
    float xr[8];
#pragma unroll
    for (int r = 0; r < 8; ++r)
        xr[r] = x[(m0 + r + lhi * 8) * T_DIM + t];

#pragma unroll
    for (int j = 0; j < 4; ++j) {
        const int col = n_strip * 64 + j * 16 + lrow;   // C/D: N = lane&15
        const float wx = Whx[col];
        const float bb = bh[col];
#pragma unroll
        for (int r = 0; r < 8; ++r) {
            const int mrow = m0 + r + lhi * 8;          // C/D: M = r + 8*(lane>=16)
            float v = acc[j][r] + xr[r] * wx + bb;
            hnext[mrow * H_DIM + col] = f32_to_bf16_rne(tanhf(v));
        }
    }
}

// ---------------------------------------------------------------------------
// Final projection: out[b,o] = h_final[b,:] . W_ph[:,o] + b_o[o]   (tiny)
// ---------------------------------------------------------------------------
__global__ void final_proj_kernel(const unsigned short* __restrict__ hfin,
                                  const float* __restrict__ Wph,
                                  const float* __restrict__ bo,
                                  float* __restrict__ out) {
    int id = blockIdx.x * blockDim.x + threadIdx.x;
    if (id >= B_DIM * O_DIM) return;
    int b = id / O_DIM, o = id % O_DIM;
    float s = bo[o];
    const unsigned short* hr = hfin + b * H_DIM;
    for (int h = 0; h < H_DIM; ++h)
        s += bf16_to_f32(hr[h]) * Wph[h * O_DIM + o];
    out[id] = s;
}

extern "C" void kernel_launch(void* const* d_in, const int* in_sizes, int n_in,
                              void* d_out, int out_size, void* d_ws, size_t ws_size,
                              hipStream_t stream) {
    const float* x   = (const float*)d_in[0];   // [256,512]
    const float* Whx = (const float*)d_in[1];   // [1,1024]
    const float* Whh = (const float*)d_in[2];   // [1024,1024]
    const float* Wph = (const float*)d_in[3];   // [1024,10]
    const float* bh  = (const float*)d_in[4];   // [1024]
    const float* bo  = (const float*)d_in[5];   // [10]
    float* out = (float*)d_out;                 // [256,10]

    unsigned char* ws = (unsigned char*)d_ws;
    unsigned int*   Bp = (unsigned int*)ws;                              // 2 MB packed W_hh
    unsigned short* h0 = (unsigned short*)(ws + (2u << 20));             // 512 KB bf16 h
    unsigned short* h1 = (unsigned short*)(ws + (2u << 20) + (512u << 10));

    // Pack weights (every call; deterministic, no cross-call state).
    pack_whh_kernel<<<2048, 256, 0, stream>>>(Whh, Bp);
    // h(0) = 0
    zero_u32_kernel<<<512, 256, 0, stream>>>((unsigned int*)h0,
                                             (B_DIM * H_DIM) / 2);

    unsigned short* bufs[2] = {h0, h1};
    for (int t = 0; t < T_DIM; ++t) {
        rnn_step_kernel<<<32, 256, 0, stream>>>(bufs[t & 1], Bp, x, Whx, bh, t,
                                                bufs[(t + 1) & 1]);
    }
    // T=512 is even -> final hidden state lives in bufs[0].
    final_proj_kernel<<<(B_DIM * O_DIM + 255) / 256, 256, 0, stream>>>(
        bufs[0], Wph, bo, out);
}
